// Attention_14353780703687
// MI455X (gfx1250) — compile-verified, matrix operand order checked by reading, not run on hardware
//
#include <hip/hip_runtime.h>
#include <hip/hip_bf16.h>

// ---------------------------------------------------------------------------
// CDNA5 (gfx1250) bf16 WMMA attention pipeline.
// - v_wmma_f32_16x16x32_bf16 for all matmul phases
// - async global->LDS staging (GLOBAL_LOAD_ASYNC_TO_LDS_B128) where contiguous
// - ds_load_b128 fragment gathers, register-resident O accumulators
// ---------------------------------------------------------------------------

typedef __bf16 bf16;
typedef __attribute__((ext_vector_type(16))) __bf16 v16bf;
typedef __attribute__((ext_vector_type(8)))  __bf16 v8bf;
typedef __attribute__((ext_vector_type(4)))  __bf16 v4bf;
typedef __attribute__((ext_vector_type(8)))  float  v8f;
typedef __attribute__((ext_vector_type(4)))  float  v4f;
typedef __attribute__((ext_vector_type(4)))  int    v4i;

#define WMMA_BF16(a, b, c) \
  __builtin_amdgcn_wmma_f32_16x16x32_bf16(false, (a), false, (b), (short)0, (c), false, false)

// ---- async global -> LDS (16B per lane), with synchronous fallback ---------
#if defined(__has_builtin)
#if __has_builtin(__builtin_amdgcn_global_load_async_to_lds_b128)
#define ASYNC_LDS 1
#endif
#endif
#ifndef ASYNC_LDS
#define ASYNC_LDS 0
#endif

typedef __attribute__((address_space(1))) v4i g_v4i;
typedef __attribute__((address_space(3))) v4i l_v4i;

__device__ __forceinline__ void copy16_to_lds(bf16* lds_dst, const bf16* gsrc) {
#if ASYNC_LDS
  bf16* gs = const_cast<bf16*>(gsrc);
  __builtin_amdgcn_global_load_async_to_lds_b128(
      (g_v4i*)gs, (l_v4i*)lds_dst, 0, 0);
#else
  *(v8bf*)lds_dst = *(const v8bf*)gsrc;
#endif
}

__device__ __forceinline__ void async_join() {
#if ASYNC_LDS
#if __has_builtin(__builtin_amdgcn_s_wait_asynccnt)
  __builtin_amdgcn_s_wait_asynccnt(0);
#else
  asm volatile("s_wait_asynccnt 0" ::: "memory");
#endif
#endif
}

// Fragment load: per ISA 7.12.2, lane<16 holds row=lane, K={0..7,16..23};
// lane>=16 holds row=lane-16, K={8..15,24..31}. With K-contiguous LDS tiles
// this is two ds_load_b128 per fragment. Works for A tiles (row-major [M][K])
// and for B tiles staged transposed ([N][K]).
__device__ __forceinline__ v16bf frag_ld(const bf16* t, int ld) {
  const int lane = threadIdx.x & 31;
  const int row  = lane & 15;
  const int kb   = (lane & 16) ? 8 : 0;
  const bf16* p  = t + row * ld + kb;
  v8bf lo = *(const v8bf*)(p);
  v8bf hi = *(const v8bf*)(p + 16);
  return __builtin_shufflevector(lo, hi, 0, 1, 2, 3, 4, 5, 6, 7,
                                 8, 9, 10, 11, 12, 13, 14, 15);
}

// ---------------------------------------------------------------------------
// Kernel 0: fp32 -> bf16 conversions + head-split re-layout of mem_k/mem_v.
// ---------------------------------------------------------------------------
__device__ __forceinline__ void cvt4(bf16* __restrict__ dst,
                                     const float* __restrict__ src) {
  v4f s = *(const v4f*)src;
  v4bf o;
#pragma unroll
  for (int u = 0; u < 4; ++u) o[u] = (bf16)s[u];
  *(v4bf*)dst = o;
}

__global__ void convert_kernel(const float* __restrict__ x,
                               const float* __restrict__ Wq,
                               const float* __restrict__ Wkv,
                               const float* __restrict__ Wo,
                               const float* __restrict__ mk,
                               const float* __restrict__ mv,
                               bf16* __restrict__ xh,  bf16* __restrict__ wqh,
                               bf16* __restrict__ wkvh, bf16* __restrict__ woh,
                               bf16* __restrict__ kh,  bf16* __restrict__ vh) {
  const long long total4 = 2621440LL;  // 10485760 elements / 4
  for (long long c = (long long)blockIdx.x * blockDim.x + threadIdx.x;
       c < total4; c += (long long)gridDim.x * blockDim.x) {
    long long r = c;
    if (r < 524288LL) { cvt4(xh + r * 4, x + r * 4); continue; }
    r -= 524288LL;
    if (r < 262144LL) { cvt4(wqh + r * 4, Wq + r * 4); continue; }
    r -= 262144LL;
    if (r < 524288LL) { cvt4(wkvh + r * 4, Wkv + r * 4); continue; }
    r -= 524288LL;
    if (r < 262144LL) { cvt4(woh + r * 4, Wo + r * 4); continue; }
    r -= 262144LL;
    if (r < 524288LL) {  // mem_k [b, j, h*64+d] -> kh [b, h, j, d]
      long long e = r * 4;
      int b = (int)(e >> 20), rem = (int)(e & 1048575LL);
      int j = rem >> 10, hd = rem & 1023, h = hd >> 6, d = hd & 63;
      cvt4(kh + ((((long long)(b * 16 + h)) * 2048 + j) * 64 + d), mk + e);
      continue;
    }
    r -= 524288LL;
    {
      long long e = r * 4;
      int b = (int)(e >> 20), rem = (int)(e & 1048575LL);
      int j = rem >> 10, hd = rem & 1023, h = hd >> 6, d = hd & 63;
      cvt4(vh + ((((long long)(b * 16 + h)) * 2048 + j) * 64 + d), mv + e);
    }
  }
}

// ---------------------------------------------------------------------------
// Shared 64x64 GEMM tile core: 128 threads = 4 waves; wave w owns rows
// [16w,16w+16) x 64 cols. K stepped by 32; A staged row-major (async),
// B staged TRANSPOSED ([n][k]) through VGPRs.
// ---------------------------------------------------------------------------
__device__ __forceinline__ void gemm64x64_core(const bf16* __restrict__ A, int lda,
                                               const bf16* __restrict__ B, int ldb,
                                               int K, v8f acc[4]) {
  __shared__ alignas(16) bf16 As[64 * 32];   // [m][k]
  __shared__ alignas(16) bf16 Bts[64 * 32];  // [n][k]
  const int t = threadIdx.x;
  const int w = t >> 5;
#pragma unroll
  for (int nt = 0; nt < 4; ++nt) acc[nt] = (v8f){0.f, 0.f, 0.f, 0.f, 0.f, 0.f, 0.f, 0.f};

  for (int k0 = 0; k0 < K; k0 += 32) {
    // A tile 64x32: 2x 16B async copies per thread (no bounds check)
#pragma unroll
    for (int it = 0; it < 2; ++it) {
      int e = t + it * 128;
      int r = e >> 2, cc = (e & 3) * 8;
      copy16_to_lds(&As[r * 32 + cc], A + (long long)r * lda + k0 + cc);
    }
    // B tile 32x64: coalesced 16B reads, transpose into Bts on LDS write
#pragma unroll
    for (int it = 0; it < 2; ++it) {
      int e = t + it * 128;
      int k = e >> 3, n = (e & 7) * 8;
      v8bf vv = *(const v8bf*)(B + (long long)(k0 + k) * ldb + n);
#pragma unroll
      for (int u = 0; u < 8; ++u) Bts[(n + u) * 32 + k] = vv[u];
    }
    async_join();
    __syncthreads();
    v16bf a = frag_ld(&As[16 * w * 32], 32);
#pragma unroll
    for (int nt = 0; nt < 4; ++nt) {
      v16bf b = frag_ld(&Bts[16 * nt * 32], 32);
      acc[nt] = WMMA_BF16(a, b, acc[nt]);
    }
    __syncthreads();
  }
}

// ---- q = (x @ Wq) * scale, head-split -> qh [b, h, i, d] (bf16) -------------
__global__ void gemm_q_kernel(const bf16* __restrict__ xh,
                              const bf16* __restrict__ wqh,
                              bf16* __restrict__ qh) {
  const int i0 = blockIdx.y * 64, n0 = blockIdx.x * 64;
  v8f acc[4];
  gemm64x64_core(xh + (long long)i0 * 1024, 1024, wqh + n0, 1024, 1024, acc);
  const int lane = threadIdx.x & 31, w = threadIdx.x >> 5;
  const int col = lane & 15, rb = (lane & 16) ? 8 : 0;
#pragma unroll
  for (int nt = 0; nt < 4; ++nt)
#pragma unroll
    for (int v = 0; v < 8; ++v) {
      int r = i0 + 16 * w + rb + v;           // flat row = b*1024 + i
      int c = n0 + 16 * nt + col;             // inner col = h*64 + d
      int bb = r >> 10, ii = r & 1023, h = c >> 6, dd = c & 63;
      qh[(((long long)(bb * 16 + h)) * 1024 + ii) * 64 + dd] =
          (bf16)(acc[nt][v] * 0.125f);        // SCALE = 64^-0.5
    }
}

// ---- [self_k | self_v] = x @ Wkv, head-split into kh/vh at j offset 1024 ----
__global__ void gemm_kv_kernel(const bf16* __restrict__ xh,
                               const bf16* __restrict__ wkvh,
                               bf16* __restrict__ kh, bf16* __restrict__ vh) {
  const int i0 = blockIdx.y * 64, n0 = blockIdx.x * 64;
  v8f acc[4];
  gemm64x64_core(xh + (long long)i0 * 1024, 1024, wkvh + n0, 2048, 1024, acc);
  const int lane = threadIdx.x & 31, w = threadIdx.x >> 5;
  const int col = lane & 15, rb = (lane & 16) ? 8 : 0;
#pragma unroll
  for (int nt = 0; nt < 4; ++nt)
#pragma unroll
    for (int v = 0; v < 8; ++v) {
      int r = i0 + 16 * w + rb + v;
      int c = n0 + 16 * nt + col;             // 0..2047: k then v
      int bb = r >> 10, ii = r & 1023;
      bf16 val = (bf16)acc[nt][v];
      if (c < 1024) {
        int h = c >> 6, dd = c & 63;
        kh[(((long long)(bb * 16 + h)) * 2048 + 1024 + ii) * 64 + dd] = val;
      } else {
        int cv = c - 1024, h = cv >> 6, dd = cv & 63;
        vh[(((long long)(bb * 16 + h)) * 2048 + 1024 + ii) * 64 + dd] = val;
      }
    }
}

// ---------------------------------------------------------------------------
// Flash-attention kernel: block = 128 threads / 4 waves handles 64 query rows
// of one (b,h). O accumulators live in registers; online softmax over 64-wide
// j tiles with causal mask + pos_bias (read once, f32x4, next tile prefetched)
// ---------------------------------------------------------------------------
__global__ void attn_kernel(const bf16* __restrict__ qh,
                            const bf16* __restrict__ kh,
                            const bf16* __restrict__ vh,
                            const float* __restrict__ pos_bias,
                            bf16* __restrict__ attn_out) {
  __shared__ alignas(16) bf16  Qs[64 * 64];    // [i][d]
  __shared__ alignas(16) bf16  Ks[64 * 64];    // [j][d]  (B^T for QK^T)
  __shared__ alignas(16) bf16  Vts[64 * 64];   // [d][j]  (B^T for P.V)
  __shared__ alignas(16) float SsT[64 * 64];   // [j][i]
  __shared__ alignas(16) bf16  Ps[64 * 64];    // [i][j]
  __shared__ float row_m[64], row_l[64], row_scale[64];

  const int t = threadIdx.x, w = t >> 5, lane = t & 31;
  const int col = lane & 15, rb = (lane & 16) ? 8 : 0;
  const int i0 = blockIdx.x * 64;
  const int bh = blockIdx.y;              // b*16 + h
  const int h  = bh & 15;

  v8f oacc[4];
#pragma unroll
  for (int dt = 0; dt < 4; ++dt) oacc[dt] = (v8f){0.f, 0.f, 0.f, 0.f, 0.f, 0.f, 0.f, 0.f};

  const bf16* qbase = qh + ((long long)bh * 1024 + i0) * 64;
#pragma unroll
  for (int it = 0; it < 4; ++it) {
    int e = t + it * 128;
    copy16_to_lds(&Qs[e * 8], qbase + e * 8);
  }
  if (t < 64) { row_m[t] = -3.0e38f; row_l[t] = 0.0f; }
  async_join();
  __syncthreads();

  const int jtiles = (i0 >> 6) + 17;      // cover j in [0, i0+63+1024]
  for (int jt = 0; jt < jtiles; ++jt) {
    const int j0 = jt * 64;
    const bf16* kbase = kh + ((long long)bh * 2048 + j0) * 64;
    const bf16* vbase = vh + ((long long)bh * 2048 + j0) * 64;
#pragma unroll
    for (int it = 0; it < 4; ++it) {       // K: contiguous async copies
      int e = t + it * 128;
      copy16_to_lds(&Ks[e * 8], kbase + e * 8);
    }
#pragma unroll
    for (int it = 0; it < 4; ++it) {       // V: transpose into Vts[d][j]
      int e = t + it * 128;
      v8bf vv = *(const v8bf*)(vbase + e * 8);
      int j = e >> 3, dd = (e & 7) * 8;
#pragma unroll
      for (int u = 0; u < 8; ++u) Vts[(dd + u) * 64 + j] = vv[u];
    }
    async_join();
    __syncthreads();

    // ---- S = Q K^T : wave w rows [16w,16w+16) x 64 j-cols; store S^T ------
#pragma unroll
    for (int nt = 0; nt < 4; ++nt) {
      v8f sa = (v8f){0.f, 0.f, 0.f, 0.f, 0.f, 0.f, 0.f, 0.f};
#pragma unroll
      for (int ks = 0; ks < 2; ++ks) {
        v16bf a = frag_ld(&Qs[16 * w * 64 + 32 * ks], 64);
        v16bf b = frag_ld(&Ks[16 * nt * 64 + 32 * ks], 64);
        sa = WMMA_BF16(a, b, sa);
      }
      float* p = &SsT[(16 * nt + col) * 64 + 16 * w + rb];
      *(v4f*)(p)     = (v4f){sa[0], sa[1], sa[2], sa[3]};
      *(v4f*)(p + 4) = (v4f){sa[4], sa[5], sa[6], sa[7]};
    }
    __syncthreads();

    // ---- online softmax: thread t owns row i = i0+t -----------------------
    if (t < 64) {
      const int ig = i0 + t;
      const float* pbp = pos_bias + ((size_t)h * 1024 + ig) * 2048 + j0;
      if (jt + 1 < jtiles) __builtin_prefetch(pbp + 64, 0, 1);  // next j tile
      float mold = row_m[t], mnew = mold;
#pragma unroll
      for (int q4 = 0; q4 < 16; ++q4) {
        v4f pv = *(const v4f*)(pbp + q4 * 4);
#pragma unroll
        for (int u = 0; u < 4; ++u) {
          int jj = q4 * 4 + u;
          float vsc = SsT[jj * 64 + t] + pv[u];
          if (j0 + jj > ig + 1024) vsc = -3.0e38f;   // causal mask
          SsT[jj * 64 + t] = vsc;
          mnew = fmaxf(mnew, vsc);
        }
      }
      float sc = __expf(mold - mnew);
      float l  = row_l[t] * sc;
      for (int jj = 0; jj < 64; ++jj) {
        float p = __expf(SsT[jj * 64 + t] - mnew);
        l += p;
        Ps[t * 64 + jj] = (bf16)p;
      }
      row_m[t] = mnew; row_l[t] = l; row_scale[t] = sc;
    }
    __syncthreads();

    // ---- rescale register O by exp(m_old - m_new) -------------------------
    float scv[8];
#pragma unroll
    for (int v = 0; v < 8; ++v) scv[v] = row_scale[16 * w + rb + v];
#pragma unroll
    for (int dt = 0; dt < 4; ++dt)
#pragma unroll
      for (int v = 0; v < 8; ++v) oacc[dt][v] *= scv[v];

    // ---- O += P @ V -------------------------------------------------------
#pragma unroll
    for (int ks = 0; ks < 2; ++ks) {
      v16bf a = frag_ld(&Ps[16 * w * 64 + 32 * ks], 64);
#pragma unroll
      for (int dt = 0; dt < 4; ++dt) {
        v16bf b = frag_ld(&Vts[16 * dt * 64 + 32 * ks], 64);
        oacc[dt] = WMMA_BF16(a, b, oacc[dt]);
      }
    }
    __syncthreads();
  }

  // ---- normalize and write merged-head layout [b*1024+i, h*64+d] (bf16) ---
  const int bb = bh >> 4;
  float lv[8];
#pragma unroll
  for (int v = 0; v < 8; ++v) lv[v] = 1.0f / row_l[16 * w + rb + v];
#pragma unroll
  for (int dt = 0; dt < 4; ++dt)
#pragma unroll
    for (int v = 0; v < 8; ++v) {
      int r = 16 * w + rb + v;
      int d = 16 * dt + col;
      attn_out[((size_t)(bb * 1024 + i0 + r)) * 1024 + h * 64 + d] =
          (bf16)(oacc[dt][v] * lv[v]);
    }
}

// ---- out = attn @ Wo + bo (fp32 result) ------------------------------------
__global__ void gemm_out_kernel(const bf16* __restrict__ at,
                                const bf16* __restrict__ woh,
                                const float* __restrict__ bo,
                                float* __restrict__ out) {
  const int i0 = blockIdx.y * 64, n0 = blockIdx.x * 64;
  v8f acc[4];
  gemm64x64_core(at + (long long)i0 * 1024, 1024, woh + n0, 1024, 1024, acc);
  const int lane = threadIdx.x & 31, w = threadIdx.x >> 5;
  const int col = lane & 15, rb = (lane & 16) ? 8 : 0;
#pragma unroll
  for (int nt = 0; nt < 4; ++nt)
#pragma unroll
    for (int v = 0; v < 8; ++v) {
      int r = i0 + 16 * w + rb + v;
      int c = n0 + 16 * nt + col;
      out[(long long)r * 1024 + c] = acc[nt][v] + bo[c];
    }
}

// ---------------------------------------------------------------------------
extern "C" void kernel_launch(void* const* d_in, const int* in_sizes, int n_in,
                              void* d_out, int out_size, void* d_ws, size_t ws_size,
                              hipStream_t stream) {
  (void)in_sizes; (void)n_in; (void)out_size; (void)ws_size;
  const float* x   = (const float*)d_in[0];
  const float* mk  = (const float*)d_in[1];
  const float* mv  = (const float*)d_in[2];
  const float* pb  = (const float*)d_in[3];
  const float* Wq  = (const float*)d_in[4];
  const float* Wkv = (const float*)d_in[5];
  const float* Wo  = (const float*)d_in[6];
  const float* bo  = (const float*)d_in[7];
  float* out = (float*)d_out;

  // Workspace layout (bf16 elements): ~36 MB total
  bf16* ws   = (bf16*)d_ws;
  bf16* xh   = ws + 0;          // x            [2048, 1024]         2M
  bf16* wqh  = ws + 2097152;    // Wq           [1024, 1024]         1M
  bf16* wkvh = ws + 3145728;    // Wkv          [1024, 2048]         2M
  bf16* woh  = ws + 5242880;    // Wo           [1024, 1024]         1M
  bf16* qh   = ws + 6291456;    // q split      [2, 16, 1024, 64]    2M
  bf16* kh   = ws + 8388608;    // k concat     [2, 16, 2048, 64]    4M
  bf16* vh   = ws + 12582912;   // v concat     [2, 16, 2048, 64]    4M
  bf16* at   = ws + 16777216;   // attn output  [2048, 1024]         2M

  convert_kernel<<<4096, 256, 0, stream>>>(x, Wq, Wkv, Wo, mk, mv,
                                           xh, wqh, wkvh, woh, kh, vh);
  gemm_q_kernel <<<dim3(16, 32), 128, 0, stream>>>(xh, wqh, qh);
  gemm_kv_kernel<<<dim3(32, 32), 128, 0, stream>>>(xh, wkvh, kh, vh);
  attn_kernel   <<<dim3(16, 32), 128, 0, stream>>>(qh, kh, vh, pb, at);
  gemm_out_kernel<<<dim3(16, 32), 128, 0, stream>>>(at, woh, bo, out);
}
